// SelfAttentionWithRoPE_52879637348738
// MI455X (gfx1250) — matrix operand order repaired and checked by reading, not compiled, not run on hardware
//
#include <hip/hip_runtime.h>

#define DEVINL __device__ __forceinline__

typedef __attribute__((ext_vector_type(16))) __bf16 v16bf;
typedef __attribute__((ext_vector_type(8)))  __bf16 v8bf;
typedef __attribute__((ext_vector_type(8)))  float  v8f;
typedef __attribute__((ext_vector_type(4)))  float  f32x4;
typedef __attribute__((ext_vector_type(4)))  unsigned int u32x4;
typedef __attribute__((ext_vector_type(8)))  int i32x8;
typedef __attribute__((ext_vector_type(4)))  int i32x4;

static constexpr int B_  = 2;
static constexpr int N_  = 2048;
static constexpr int F_  = 2048;
static constexpr int H_  = 16;
static constexpr int DH_ = 128;
static constexpr int M_  = B_ * N_;   // 4096 token rows
static constexpr int F3_ = 3 * F_;    // 6144

// ---------- native bf16 conversions ----------
DEVINL __bf16 f2bf(float f) { return (__bf16)f; }
DEVINL float  bf2f(__bf16 b) { return (float)b; }

DEVINL v16bf cat8(v8bf lo, v8bf hi) {
  return __builtin_shufflevector(lo, hi, 0, 1, 2, 3, 4, 5, 6, 7, 8, 9, 10, 11, 12,
                                 13, 14, 15);
}

// ---------- WMMA wrapper ----------
DEVINL v8f wmma_bf16(v16bf a, v16bf b, v8f c) {
  return __builtin_amdgcn_wmma_f32_16x16x32_bf16(false, a, false, b, (short)0, c,
                                                 false, false);
}

// =====================================================================
// Tensor Data Mover: 2D tile global -> LDS with LDS row padding.
// D# per CDNA5 ISA 8.3/8.4. This toolchain: 6-arg builtin.
// =====================================================================
DEVINL void tdm_load_2d(unsigned lds_off, const void* gptr,
                        unsigned ds_log,          // 0=1B,1=2B,2=4B
                        unsigned tile0, unsigned tile1,
                        unsigned long long stride0_elems,
                        unsigned pad_interval_code, unsigned pad_amount_code,
                        unsigned pad_enable) {
  unsigned long long ga = (unsigned long long)(uintptr_t)gptr;
  u32x4 g0;
  g0[0] = 1u;                                          // count=1, user mode
  g0[1] = lds_off;                                     // LDS byte address
  g0[2] = (unsigned)(ga & 0xffffffffu);                // global_addr[31:0]
  g0[3] = (unsigned)((ga >> 32) & 0x01ffffffu) | (2u << 30);  // addr[56:32]|type=2
  i32x8 g1;
  g1[0] = (int)((ds_log << 16) | (pad_enable << 20) |
                (pad_interval_code << 22) | (pad_amount_code << 25));
  g1[1] = (int)((tile0 & 0xffffu) << 16);              // tensor_dim0[15:0]
  g1[2] = (int)(((tile0 >> 16) & 0xffffu) | ((tile1 & 0xffffu) << 16));
  g1[3] = (int)(((tile1 >> 16) & 0xffffu) | ((tile0 & 0xffffu) << 16)); // tile_dim0
  g1[4] = (int)(tile1 & 0xffffu);                      // tile_dim1 (tile_dim2=0)
  g1[5] = (int)(stride0_elems & 0xffffffffu);          // tensor_dim0_stride lo
  g1[6] = (int)((stride0_elems >> 32) & 0xffffu);      // stride hi; dim1_stride=0
  g1[7] = 0;
  i32x4 z4 = {0, 0, 0, 0};
  i32x8 z8 = {0, 0, 0, 0, 0, 0, 0, 0};
  __builtin_amdgcn_tensor_load_to_lds(g0, g1, z4, z4, z8, 0);
}

// ---------- fragment loaders (all addresses 16B-aligned by construction) ----
// A 16x32 (bf16): lane row = row0+lane%16; e<8 -> K=kh+e, e>=8 -> K=16+kh+(e-8).
DEVINL v16bf a_frag_g_bf16(const __bf16* __restrict__ base, int ld, int row0,
                           int k0, int lane) {
  const __bf16* p =
      base + (size_t)(row0 + (lane & 15)) * ld + k0 + ((lane & 16) ? 8 : 0);
  v8bf lo = *(const v8bf*)p;
  v8bf hi = *(const v8bf*)(p + 16);
  return cat8(lo, hi);
}
DEVINL v16bf a_frag_lds_bf16(const __bf16* p0, int stride, int row0, int k0,
                             int lane) {
  const __bf16* p =
      p0 + (size_t)(row0 + (lane & 15)) * stride + k0 + ((lane & 16) ? 8 : 0);
  v8bf lo = *(const v8bf*)p;
  v8bf hi = *(const v8bf*)(p + 16);
  return cat8(lo, hi);
}
// B 32x16 with logical B[k][n] stored as 16 contiguous K per "n-row":
// lane col = n0+lane%16; e<8 -> K=kh+e, e>=8 -> K=16+kh+(e-8).
DEVINL v16bf bT_frag_lds_bf16(const __bf16* p0, int stride, int n0, int k0,
                              int lane) {
  const __bf16* p =
      p0 + (size_t)(n0 + (lane & 15)) * stride + k0 + ((lane & 16) ? 8 : 0);
  v8bf lo = *(const v8bf*)p;
  v8bf hi = *(const v8bf*)(p + 16);
  return cat8(lo, hi);
}

// =====================================================================
// Kernel 0: one-shot f32 -> bf16 conversion (vectorized, memory bound).
// =====================================================================
__global__ __launch_bounds__(256) void cvt_f32_bf16(const float* __restrict__ in,
                                                    __bf16* __restrict__ outp) {
  const int i = blockIdx.x * blockDim.x + threadIdx.x;  // one v8 per thread
  f32x4 a = ((const f32x4*)in)[2 * i];
  f32x4 b = ((const f32x4*)in)[2 * i + 1];
  v8bf o;
#pragma unroll
  for (int e = 0; e < 4; ++e) { o[e] = f2bf(a[e]); o[e + 4] = f2bf(b[e]); }
  ((v8bf*)outp)[i] = o;
}

// =====================================================================
// Shared GEMM body: C[M x Ncols] = A(bf16) @ Bw(bf16,row-major-over-n)^T.
// Block 256thr/8 waves, tile 128x256, K-step 64.
// TDM double-buffers bf16 tiles; rows 128B padded to 144B (stride 72 elems).
// =====================================================================
static constexpr int G_A_BYTES = 128 * 72 * 2;  // 18432
static constexpr int G_B_BYTES = 256 * 72 * 2;  // 36864
static constexpr int G_SMEM = 2 * G_A_BYTES + 2 * G_B_BYTES;  // 110592

template <typename EPI>
DEVINL void gemm_body(const __bf16* __restrict__ A, const __bf16* __restrict__ Bw,
                      char* smem, int ldA, int ldB, EPI&& epilogue) {
  const int lane = threadIdx.x & 31;
  const int wv = threadIdx.x >> 5;
  const int rowBase = blockIdx.y * 128;
  const int colBase = blockIdx.x * 256;
  const int rloc = (wv >> 2) * 64;
  const int cloc = (wv & 3) * 64;

  const unsigned offA[2] = {0u, (unsigned)G_A_BYTES};
  const unsigned offB[2] = {(unsigned)(2 * G_A_BYTES),
                            (unsigned)(2 * G_A_BYTES + G_B_BYTES)};

  if (wv == 0) {
    tdm_load_2d(offA[0], A + (size_t)rowBase * ldA, 1, 64, 128, ldA, 4, 3, 1);
    tdm_load_2d(offB[0], Bw + (size_t)colBase * ldB, 1, 64, 256, ldB, 4, 3, 1);
  }

  v8f acc[4][4] = {};
  int buf = 0;
  for (int k0 = 0; k0 < F_; k0 += 64) {
    if (wv == 0) {
      if (k0 + 64 < F_) {
        tdm_load_2d(offA[buf ^ 1], A + (size_t)rowBase * ldA + k0 + 64, 1, 64, 128,
                    ldA, 4, 3, 1);
        tdm_load_2d(offB[buf ^ 1], Bw + (size_t)colBase * ldB + k0 + 64, 1, 64, 256,
                    ldB, 4, 3, 1);
        __builtin_amdgcn_s_wait_tensorcnt(2);  // oldest pair (current buf) done
      } else {
        __builtin_amdgcn_s_wait_tensorcnt(0);
      }
    }
    __syncthreads();
    const __bf16* ldsA = (const __bf16*)(smem + offA[buf]);
    const __bf16* ldsB = (const __bf16*)(smem + offB[buf]);
#pragma unroll
    for (int kk = 0; kk < 2; ++kk) {
      v16bf a[4];
#pragma unroll
      for (int i = 0; i < 4; ++i)
        a[i] = a_frag_lds_bf16(ldsA, 72, rloc + 16 * i, kk * 32, lane);
#pragma unroll
      for (int j = 0; j < 4; ++j) {
        v16bf b = bT_frag_lds_bf16(ldsB, 72, cloc + 16 * j, kk * 32, lane);
#pragma unroll
        for (int i = 0; i < 4; ++i) acc[i][j] = wmma_bf16(a[i], b, acc[i][j]);
      }
    }
    __syncthreads();
    buf ^= 1;
  }

  const int half = lane >> 4;
#pragma unroll
  for (int j = 0; j < 4; ++j) {
    const int col = colBase + cloc + 16 * j + (lane & 15);
#pragma unroll
    for (int i = 0; i < 4; ++i)
#pragma unroll
      for (int r = 0; r < 8; ++r) {
        const int row = rowBase + rloc + 16 * i + r + 8 * half;
        epilogue(row, col, acc[i][j][r]);
      }
  }
}

// Kernel 1: qkv = x_bf16 @ Wqkv_bf16^T + b -> bf16 ws
__global__ __launch_bounds__(256) void qkv_gemm(const __bf16* __restrict__ xb,
                                                const __bf16* __restrict__ Wqkvb,
                                                const float* __restrict__ bqkv,
                                                __bf16* __restrict__ qkv) {
  extern __shared__ char smem[];
  gemm_body(xb, Wqkvb, smem, F_, F_, [&](int row, int col, float v) {
    qkv[(size_t)row * F3_ + col] = f2bf(v + bqkv[col]);
  });
}

// Kernel 4: out = y_bf16 @ Wout_bf16^T + b -> f32 output
__global__ __launch_bounds__(256) void out_gemm(const __bf16* __restrict__ yb,
                                                const __bf16* __restrict__ Woutb,
                                                const float* __restrict__ bout,
                                                float* __restrict__ out) {
  extern __shared__ char smem[];
  gemm_body(yb, Woutb, smem, F_, F_, [&](int row, int col, float v) {
    out[(size_t)row * F_ + col] = v + bout[col];
  });
}

// =====================================================================
// Kernel 2: in-place RoPE on q and k halves of qkv (bf16)
// =====================================================================
__global__ __launch_bounds__(256) void rope_kernel(__bf16* __restrict__ qkv,
                                                   const int* __restrict__ pos_ids) {
  const int idx = blockIdx.x * blockDim.x + threadIdx.x;  // < 2^22
  const int j = idx & 63;
  const int h = (idx >> 6) & (H_ - 1);
  const int n = (idx >> 10) & (N_ - 1);
  const int b = idx >> 21;
  const float pos = (float)pos_ids[n];
  const float theta = __expf(-(float)j * (9.210340371976184f / 64.0f));
  float s, c;
  __sincosf(pos * theta, &s, &c);
  const size_t base = ((size_t)(b * N_ + n)) * F3_ + (size_t)h * DH_;
  {
    float x0 = bf2f(qkv[base + j]), x1 = bf2f(qkv[base + j + 64]);
    qkv[base + j] = f2bf(x0 * c - x1 * s);
    qkv[base + j + 64] = f2bf(x1 * c + x0 * s);
  }
  {
    const size_t kb = base + F_;
    float x0 = bf2f(qkv[kb + j]), x1 = bf2f(qkv[kb + j + 64]);
    qkv[kb + j] = f2bf(x0 * c - x1 * s);
    qkv[kb + j + 64] = f2bf(x1 * c + x0 * s);
  }
}

// =====================================================================
// Kernel 3: flash attention. Block 128thr/4 waves; each wave 32 q-rows.
// Per 64-key tile: TDM stages K (64x128 bf16, rows padded 256B->272B);
// all threads transpose V into ldsVt[d][k] (stride 72 bf16);
// P staged through wave-private LDS slab to become WMMA A-fragments.
// =====================================================================
static constexpr int A_K_BYTES = 64 * 136 * 2;    // 17408 (stride 136 bf16)
static constexpr int A_VT_BYTES = 128 * 72 * 2;   // 18432 (stride 72 bf16)
static constexpr int A_P_BYTES = 4 * 32 * 64 * 2; // 16384
static constexpr int A_SMEM = A_K_BYTES + A_VT_BYTES + A_P_BYTES;  // 52224

__global__ __launch_bounds__(128) void attn_kernel(const __bf16* __restrict__ qkv,
                                                   __bf16* __restrict__ y) {
  extern __shared__ char smem[];
  __bf16* ldsK = (__bf16*)smem;
  __bf16* ldsVt = (__bf16*)(smem + A_K_BYTES);
  const int lane = threadIdx.x & 31;
  const int wv = threadIdx.x >> 5;
  __bf16* P = (__bf16*)(smem + A_K_BYTES + A_VT_BYTES) + wv * (32 * 64);

  const int tile = blockIdx.x & 15;
  const int bh = blockIdx.x >> 4;
  const int b = bh >> 4;
  const int h = bh & (H_ - 1);
  const int row0 = tile * 128 + wv * 32;

  const __bf16* qb = qkv + (size_t)b * N_ * F3_ + (size_t)h * DH_;
  const __bf16* kb = qb + F_;
  const __bf16* vb = qb + 2 * F_;
  const float scale = 0.08838834764831845f;  // 1/sqrt(128)

  v16bf qa[2][4];
#pragma unroll
  for (int i = 0; i < 2; ++i)
#pragma unroll
    for (int kk = 0; kk < 4; ++kk)
      qa[i][kk] = a_frag_g_bf16(qb, F3_, row0 + 16 * i, kk * 32, lane);

  v8f o[2][8] = {};
  float mrow[2][8], lrow[2][8];
#pragma unroll
  for (int i = 0; i < 2; ++i)
#pragma unroll
    for (int r = 0; r < 8; ++r) { mrow[i][r] = -1e30f; lrow[i][r] = 0.0f; }

  const int half = lane >> 4;

  for (int c0 = 0; c0 < N_; c0 += 64) {
    // ---- stage K tile via TDM (wave 0); V tile transposed by all threads ----
    if (wv == 0)
      tdm_load_2d(0u, kb + (size_t)c0 * F3_, 1, 128, 64, F3_, 5, 3, 1);
    {
      const int k = threadIdx.x & 63;           // key row within tile
      const int d0 = (threadIdx.x >> 6) * 64;   // half of head dim
      const u32x4* vrow = (const u32x4*)(vb + (size_t)(c0 + k) * F3_ + d0);
#pragma unroll
      for (int q = 0; q < 8; ++q) {
        const u32x4 u4 = vrow[q];
#pragma unroll
        for (int w = 0; w < 4; ++w) {
          const unsigned u = u4[w];
          const int d = d0 + 8 * q + 2 * w;
          ldsVt[(size_t)d * 72 + k] =
              __builtin_bit_cast(__bf16, (unsigned short)(u & 0xffffu));
          ldsVt[(size_t)(d + 1) * 72 + k] =
              __builtin_bit_cast(__bf16, (unsigned short)(u >> 16));
        }
      }
    }
    if (wv == 0) __builtin_amdgcn_s_wait_tensorcnt(0);
    __syncthreads();

    // ---- S = Q @ K^T ----
    v8f s[2][4] = {};
#pragma unroll
    for (int kk = 0; kk < 4; ++kk)
#pragma unroll
      for (int j = 0; j < 4; ++j) {
        v16bf bk = bT_frag_lds_bf16(ldsK, 136, 16 * j, kk * 32, lane);
#pragma unroll
        for (int i = 0; i < 2; ++i) s[i][j] = wmma_bf16(qa[i][kk], bk, s[i][j]);
      }

    // ---- online softmax ----
#pragma unroll
    for (int i = 0; i < 2; ++i) {
#pragma unroll
      for (int r = 0; r < 8; ++r) {
        float t = fmaxf(fmaxf(s[i][0][r], s[i][1][r]), fmaxf(s[i][2][r], s[i][3][r]));
        t *= scale;
#pragma unroll
        for (int m = 1; m < 16; m <<= 1) t = fmaxf(t, __shfl_xor(t, m, 32));
        const float mnew = fmaxf(mrow[i][r], t);
        const float alpha = __expf(mrow[i][r] - mnew);
        mrow[i][r] = mnew;
        float ps = 0.0f;
#pragma unroll
        for (int j = 0; j < 4; ++j) {
          const float p = __expf(s[i][j][r] * scale - mnew);
          s[i][j][r] = p;
          ps += p;
        }
#pragma unroll
        for (int m = 1; m < 16; m <<= 1) ps += __shfl_xor(ps, m, 32);
        lrow[i][r] = lrow[i][r] * alpha + ps;
#pragma unroll
        for (int nf = 0; nf < 8; ++nf) o[i][nf][r] *= alpha;
      }
    }

    // ---- P -> wave-private LDS slab, reload as A-fragments ----
#pragma unroll
    for (int i = 0; i < 2; ++i)
#pragma unroll
      for (int j = 0; j < 4; ++j)
#pragma unroll
        for (int r = 0; r < 8; ++r)
          P[(16 * i + r + 8 * half) * 64 + 16 * j + (lane & 15)] = f2bf(s[i][j][r]);

    v16bf pa[2][2];
#pragma unroll
    for (int i = 0; i < 2; ++i)
#pragma unroll
      for (int kt = 0; kt < 2; ++kt)
        pa[i][kt] = a_frag_lds_bf16(P, 64, 16 * i, kt * 32, lane);

    // ---- O += P @ V  (B-fragments from transposed V in LDS) ----
#pragma unroll
    for (int nf = 0; nf < 8; ++nf)
#pragma unroll
      for (int kt = 0; kt < 2; ++kt) {
        v16bf vvf = bT_frag_lds_bf16(ldsVt, 72, nf * 16, kt * 32, lane);
#pragma unroll
        for (int i = 0; i < 2; ++i) o[i][nf] = wmma_bf16(pa[i][kt], vvf, o[i][nf]);
      }
    __syncthreads();  // protect ldsK/ldsVt before next tile's staging
  }

  // ---- epilogue: y = O / l ----
#pragma unroll
  for (int i = 0; i < 2; ++i)
#pragma unroll
    for (int nf = 0; nf < 8; ++nf)
#pragma unroll
      for (int r = 0; r < 8; ++r) {
        const int grow = row0 + 16 * i + r + 8 * half;
        const float val = o[i][nf][r] / lrow[i][r];
        y[((size_t)(b * N_ + grow)) * F_ + h * DH_ + 16 * nf + (lane & 15)] = f2bf(val);
      }
}

// =====================================================================
extern "C" void kernel_launch(void* const* d_in, const int* in_sizes, int n_in,
                              void* d_out, int out_size, void* d_ws, size_t ws_size,
                              hipStream_t stream) {
  (void)in_sizes; (void)n_in; (void)out_size; (void)ws_size;
  const float* x    = (const float*)d_in[0];
  const int*   pos  = (const int*)d_in[1];
  const float* Wqkv = (const float*)d_in[2];
  const float* bqkv = (const float*)d_in[3];
  const float* Wout = (const float*)d_in[4];
  const float* bout = (const float*)d_in[5];
  float* out = (float*)d_out;

  char* ws = (char*)d_ws;
  size_t off = 0;
  __bf16* qkv   = (__bf16*)(ws + off); off += (size_t)M_ * F3_ * 2;   // 50.3 MB
  __bf16* yws   = (__bf16*)(ws + off); off += (size_t)M_ * F_ * 2;    // 16.8 MB
  __bf16* xb    = (__bf16*)(ws + off); off += (size_t)M_ * F_ * 2;    // 16.8 MB
  __bf16* Wqkvb = (__bf16*)(ws + off); off += (size_t)F3_ * F_ * 2;   // 25.2 MB
  __bf16* Woutb = (__bf16*)(ws + off); off += (size_t)F_ * F_ * 2;    //  8.4 MB

  // pre-pass conversions (memory bound, ~3us total at HBM speed)
  cvt_f32_bf16<<<(M_ * F_ / 8) / 256, 256, 0, stream>>>(x, xb);
  cvt_f32_bf16<<<(F3_ * F_ / 8) / 256, 256, 0, stream>>>(Wqkv, Wqkvb);
  cvt_f32_bf16<<<(F_ * F_ / 8) / 256, 256, 0, stream>>>(Wout, Woutb);

  dim3 g1(F3_ / 256, M_ / 128);  // 24 x 32
  qkv_gemm<<<g1, 256, G_SMEM, stream>>>(xb, Wqkvb, bqkv, qkv);

  rope_kernel<<<(B_ * N_ * H_ * 64) / 256, 256, 0, stream>>>(qkv, pos);

  attn_kernel<<<B_ * H_ * (N_ / 128), 128, A_SMEM, stream>>>(qkv, yws);

  dim3 g4(F_ / 256, M_ / 128);   // 8 x 32
  out_gemm<<<g4, 256, G_SMEM, stream>>>(yws, Woutb, bout, out);
}
